// MultiHeadSelfAttention_58497454572195
// MI455X (gfx1250) — matrix operand order repaired and checked by reading, not compiled, not run on hardware
//
#include <hip/hip_runtime.h>

// ---------------------------------------------------------------------------
// Causal MHA for MI455X (gfx1250, wave32): bf16 WMMA everywhere.
// cvt/swizzle -> QKV GEMM (double-buffered pipelined staging, wmma, V written
// transposed) -> barrier-free flash attention (wmma) -> out GEMM.
// ---------------------------------------------------------------------------

typedef __attribute__((ext_vector_type(16))) __bf16        v16bf;
typedef __attribute__((ext_vector_type(8)))  float         v8f;
typedef __attribute__((ext_vector_type(4)))  unsigned int  v4u;
typedef __attribute__((ext_vector_type(4)))  float         v4f;

#define DM     1024
#define SEQ    2048
#define BATCH  4
#define NH     16
#define HDIM   64
#define MROWS  (BATCH * SEQ)   // 8192
#define NQKV   (3 * DM)        // 3072

__device__ __forceinline__ unsigned short f2bf(float f) {
  unsigned int u = __float_as_uint(f);
  u += 0x7fffu + ((u >> 16) & 1u);        // round-to-nearest-even
  return (unsigned short)(u >> 16);
}

union FragBF { v4u q[2]; v16bf v; };      // 32B: one WMMA 16-bit A/B operand

// ------------------------- conversion / swizzle ----------------------------

__global__ void k_cvt_x(const float* __restrict__ x,
                        unsigned short* __restrict__ xb, int n4) {
  int i = blockIdx.x * blockDim.x + threadIdx.x;
  int stride = gridDim.x * blockDim.x;
  for (; i < n4; i += stride) {
    v4f f = *(const v4f*)(x + (size_t)i * 4);
    unsigned short* o = xb + (size_t)i * 4;
    o[0] = f2bf(f.x); o[1] = f2bf(f.y); o[2] = f2bf(f.z); o[3] = f2bf(f.w);
  }
}

// Pre-swizzle weights into exact B-fragment order: [N/16][K/32][lane][16 bf16].
__global__ void k_swz_w(const float* __restrict__ s0, const float* __restrict__ s1,
                        const float* __restrict__ s2, unsigned short* __restrict__ dst,
                        int total) {
  int i = blockIdx.x * blockDim.x + threadIdx.x;
  int stride = gridDim.x * blockDim.x;
  for (; i < total; i += stride) {
    int frag = i >> 9, e = i & 511;
    int l = e >> 4, ee = e & 15;
    int nt = frag >> 5, kb = frag & 31;           // K/32 == 32
    int n = nt * 16 + (l & 15);
    int k = kb * 32 + ((l & 16) ? 16 : 0) + ee;
    const float* sp = (n < 1024) ? s0 : (n < 2048) ? s1 : s2;
    dst[i] = f2bf(sp[(size_t)k * 1024 + (n & 1023)]);
  }
}

__global__ void k_bias3(const float* bq, const float* bk, const float* bv,
                        float* dst) {
  int i = blockIdx.x * blockDim.x + threadIdx.x;
  if (i < 3072)
    dst[i] = (i < 1024) ? bq[i] : (i < 2048) ? bk[i - 1024] : bv[i - 2048];
}

// ------------------------------- GEMM core ---------------------------------
// C[M,N] = A[M,1024] * Bswz + bias.  WG 256 thr = 8 waves (2M x 4N),
// wave tile 64x64 (4x4 wmma), WG tile 128x256, k-step 32.
// A tile double-buffered in LDS; global load of tile kb+1 is issued before
// the WMMAs of tile kb, ds_store + one barrier per k-step close the pipe.
// mode 0: bf16 Q,K row-major; V written TRANSPOSED [b][h][d][s].
// mode 1: f32 + bias -> of.

__global__ __launch_bounds__(256) void k_gemm(
    const unsigned short* __restrict__ A, const unsigned short* __restrict__ Bswz,
    const float* __restrict__ bias,
    unsigned short* __restrict__ oq, unsigned short* __restrict__ ok2,
    unsigned short* __restrict__ ovT, float* __restrict__ of, int mode) {
  __shared__ unsigned short At[2][128 * 40];    // pitch 40 -> conflict-free

  const int t = threadIdx.x;
  const int lane = t & 31, wid = t >> 5;
  const int wm = wid >> 2, wn = wid & 3;
  const int mBase = blockIdx.y * 128;
  const int hi = (lane & 16) ? 1 : 0;

  v8f vzero = {};
  v8f acc[4][4];
#pragma unroll
  for (int i = 0; i < 4; ++i)
#pragma unroll
    for (int j = 0; j < 4; ++j) acc[i][j] = vzero;

  const int stRow = t & 127, stC = (t >> 7) * 16;
  const unsigned short* aRow = A + (size_t)(mBase + stRow) * 1024 + stC;
  unsigned short* const stDst = &At[0][0] + stRow * 40 + stC;   // buffer 0
  const int bufStride = 128 * 40;                               // elements

  // prologue: stage tile 0 into buffer 0
  *(v4u*)stDst = *(const v4u*)aRow;
  *(v4u*)(stDst + 8) = *(const v4u*)(aRow + 8);
  __syncthreads();

#pragma unroll 1
  for (int kb = 0; kb < 32; ++kb) {
    // issue next tile's global load early (latency hidden behind WMMAs)
    v4u stage, stage1;
    if (kb < 31) {
      stage = *(const v4u*)(aRow + (kb + 1) * 32);
      stage1 = *(const v4u*)(aRow + (kb + 1) * 32 + 8);
    }

    FragBF bf[4];
#pragma unroll
    for (int j = 0; j < 4; ++j) {
      int nb = blockIdx.x * 16 + wn * 4 + j;
      const v4u* bp = (const v4u*)(Bswz + (((size_t)nb * 32 + kb) << 9) + (lane << 4));
      bf[j].q[0] = bp[0];
      bf[j].q[1] = bp[1];
    }
    FragBF af[4];
    const int kb0 = hi ? 16 : 0;
    const unsigned short* cur = &At[kb & 1][0];
#pragma unroll
    for (int i = 0; i < 4; ++i) {
      int row = wm * 64 + i * 16 + (lane & 15);
      af[i].q[0] = *(const v4u*)&cur[row * 40 + kb0];
      af[i].q[1] = *(const v4u*)&cur[row * 40 + kb0 + 8];
    }
#pragma unroll
    for (int i = 0; i < 4; ++i)
#pragma unroll
      for (int j = 0; j < 4; ++j)
        acc[i][j] = __builtin_amdgcn_wmma_f32_16x16x32_bf16(
            false, af[i].v, false, bf[j].v, (short)0, acc[i][j], false, false);

    if (kb < 31) {
      *(v4u*)(stDst + ((kb + 1) & 1) * bufStride) = stage;
      *(v4u*)(stDst + ((kb + 1) & 1) * bufStride + 8) = stage1;
      __syncthreads();                          // tile kb+1 resident
    }
  }

  // Epilogue. D frag: row = v + (hi?8:0), col = lane&15.
#pragma unroll
  for (int i = 0; i < 4; ++i) {
    int rBase = mBase + wm * 64 + i * 16 + (hi ? 8 : 0);
#pragma unroll
    for (int j = 0; j < 4; ++j) {
      int c = (blockIdx.x * 16 + wn * 4 + j) * 16 + (lane & 15);
      float bvl = bias[c];
      if (mode == 0) {
        if (c < 2048) {                         // Q / K row-major bf16
          unsigned short* dst = (c < 1024) ? oq : ok2;
          int cc = c & 1023;
#pragma unroll
          for (int v = 0; v < 8; ++v)
            dst[(size_t)(rBase + v) * 1024 + cc] = f2bf(acc[i][j][v] + bvl);
        } else {                                // V transposed: [b][h][d][s]
          int d = c - 2048, h = d >> 6, dd = d & 63;
#pragma unroll
          for (int v = 0; v < 8; ++v) {
            int r = rBase + v, bi = r >> 11, s = r & 2047;
            ovT[((size_t)(bi * NH + h) * HDIM + dd) * SEQ + s] =
                f2bf(acc[i][j][v] + bvl);
          }
        }
      } else {
#pragma unroll
        for (int v = 0; v < 8; ++v)
          of[(size_t)(rBase + v) * 1024 + c] = acc[i][j][v] + bvl;
      }
    }
  }
}

// --------------------------- flash attention -------------------------------
// WG = 4 independent waves; wave owns 16 q-rows of one (b,h). No barriers:
// K fragments and V^T fragments stream straight from global (L2-resident),
// only P's D->A layout fixup routes through per-wave LDS (in-order DS).

__global__ __launch_bounds__(128) void k_attn(
    const unsigned short* __restrict__ Q, const unsigned short* __restrict__ Kb,
    const unsigned short* __restrict__ VT, unsigned short* __restrict__ attn) {
  __shared__ unsigned short Pl[4][16 * 40];

  const int t = threadIdx.x;
  const int lane = t & 31, w = t >> 5;
  const int bi = blockIdx.z, h = blockIdx.y;
  const int qRow0 = blockIdx.x * 64 + w * 16;
  const int hi8 = (lane & 16) ? 8 : 0;
  const int col = lane & 15;
  const size_t headOff = (size_t)h * HDIM;
  const size_t bOff = (size_t)bi * SEQ;
  const unsigned short* vHead = VT + (size_t)(bi * NH + h) * HDIM * SEQ;

  // Q as A fragments (2 over d)
  FragBF qf[2];
  {
    const unsigned short* qrow = Q + (bOff + qRow0 + col) * 1024 + headOff;
#pragma unroll
    for (int ds = 0; ds < 2; ++ds) {
      qf[ds].q[0] = *(const v4u*)(qrow + ds * 32 + 2 * hi8);
      qf[ds].q[1] = *(const v4u*)(qrow + ds * 32 + 2 * hi8 + 8);
    }
  }

  v8f vzero = {};
  v8f o[4];
#pragma unroll
  for (int dt = 0; dt < 4; ++dt) o[dt] = vzero;
  float rmax[8], rs[8];
#pragma unroll
  for (int v = 0; v < 8; ++v) { rmax[v] = -1e30f; rs[v] = 0.f; }

  const int nkb = (qRow0 >> 5) + 1;             // causal: cols <= qRow0+15

  for (int kbI = 0; kbI < nkb; ++kbI) {
    const int kcol = kbI * 32;

    // S = Q * K^T (B-frags straight from global K: contiguous along d)
    v8f s[2];
#pragma unroll
    for (int jt = 0; jt < 2; ++jt) {
      FragBF kf0, kf1;
      const unsigned short* krow = Kb + (bOff + kcol + jt * 16 + col) * 1024 + headOff;
      const int dof = (lane & 16) ? 16 : 0;
      kf0.q[0] = *(const v4u*)(krow + dof);
      kf0.q[1] = *(const v4u*)(krow + dof + 8);
      kf1.q[0] = *(const v4u*)(krow + 32 + dof);
      kf1.q[1] = *(const v4u*)(krow + 32 + dof + 8);
      v8f sv = __builtin_amdgcn_wmma_f32_16x16x32_bf16(
          false, qf[0].v, false, kf0.v, (short)0, vzero, false, false);
      sv = __builtin_amdgcn_wmma_f32_16x16x32_bf16(
          false, qf[1].v, false, kf1.v, (short)0, sv, false, false);
      s[jt] = sv;
    }

    // scale + causal mask (row = qRow0 + v + hi8, col = kcol + jt*16 + col)
#pragma unroll
    for (int jt = 0; jt < 2; ++jt) {
      int cg = kcol + jt * 16 + col;
#pragma unroll
      for (int v = 0; v < 8; ++v) {
        float sv = s[jt][v] * 0.125f;
        s[jt][v] = (cg > qRow0 + v + hi8) ? -1e30f : sv;
      }
    }

    // online softmax per row (row per VGPR index, replicated over 16 lanes)
#pragma unroll
    for (int v = 0; v < 8; ++v) {
      float m = fmaxf(s[0][v], s[1][v]);
#pragma unroll
      for (int off = 1; off < 16; off <<= 1) m = fmaxf(m, __shfl_xor(m, off, 32));
      float nm = fmaxf(rmax[v], m);
      float alpha = __expf(rmax[v] - nm);
      float p0 = __expf(s[0][v] - nm);
      float p1 = __expf(s[1][v] - nm);
      float ps = p0 + p1;
#pragma unroll
      for (int off = 1; off < 16; off <<= 1) ps += __shfl_xor(ps, off, 32);
      rs[v] = rs[v] * alpha + ps;
      rmax[v] = nm;
#pragma unroll
      for (int dt = 0; dt < 4; ++dt) o[dt][v] *= alpha;
      int rl = v + hi8;
      Pl[w][rl * 40 + col]      = f2bf(p0);
      Pl[w][rl * 40 + 16 + col] = f2bf(p1);
    }

    // P (D-layout in LDS) -> A fragment; same-wave DS ops are in-order
    FragBF pf;
    {
      const unsigned short* pr = &Pl[w][(lane & 15) * 40];
      pf.q[0] = *(const v4u*)(pr + 2 * hi8);
      pf.q[1] = *(const v4u*)(pr + 2 * hi8 + 8);
    }

    // O += P * V   (V^T global: contiguous along j for each d-row)
#pragma unroll
    for (int dt = 0; dt < 4; ++dt) {
      FragBF vf;
      const unsigned short* vp =
          vHead + (size_t)(dt * 16 + col) * SEQ + kcol + ((lane & 16) ? 16 : 0);
      vf.q[0] = *(const v4u*)(vp);
      vf.q[1] = *(const v4u*)(vp + 8);
      o[dt] = __builtin_amdgcn_wmma_f32_16x16x32_bf16(
          false, pf.v, false, vf.v, (short)0, o[dt], false, false);
    }
  }

  // normalize + store bf16 row-major [b*s, h*64+d]
#pragma unroll
  for (int v = 0; v < 8; ++v) {
    float inv = 1.0f / rs[v];
    size_t rowOff = (bOff + qRow0 + v + hi8) * 1024 + headOff;
#pragma unroll
    for (int dt = 0; dt < 4; ++dt)
      attn[rowOff + dt * 16 + col] = f2bf(o[dt][v] * inv);
  }
}

// -------------------------------- launch -----------------------------------

extern "C" void kernel_launch(void* const* d_in, const int* in_sizes, int n_in,
                              void* d_out, int out_size, void* d_ws, size_t ws_size,
                              hipStream_t stream) {
  const float* x  = (const float*)d_in[0];
  const float* wq = (const float*)d_in[1];
  const float* bq = (const float*)d_in[2];
  const float* wk = (const float*)d_in[3];
  const float* bk = (const float*)d_in[4];
  const float* wv = (const float*)d_in[5];
  const float* bv = (const float*)d_in[6];
  const float* wo = (const float*)d_in[7];
  const float* bo = (const float*)d_in[8];
  float* out = (float*)d_out;

  char* p = (char*)d_ws;
  auto carve = [&](size_t bytes) {
    char* r = p; p += (bytes + 255) & ~(size_t)255; return r;
  };
  unsigned short* xb   = (unsigned short*)carve((size_t)MROWS * DM * 2);
  unsigned short* wqkv = (unsigned short*)carve((size_t)NQKV * DM * 2);
  unsigned short* wob  = (unsigned short*)carve((size_t)DM * DM * 2);
  float*          bqkv = (float*)carve(3072 * 4);
  unsigned short* Qb   = (unsigned short*)carve((size_t)MROWS * DM * 2);
  unsigned short* Kbf  = (unsigned short*)carve((size_t)MROWS * DM * 2);
  unsigned short* VbT  = (unsigned short*)carve((size_t)MROWS * DM * 2);
  unsigned short* Ab   = (unsigned short*)carve((size_t)MROWS * DM * 2);

  k_cvt_x<<<4096, 256, 0, stream>>>(x, xb, MROWS * DM / 4);
  k_swz_w<<<6144, 256, 0, stream>>>(wq, wk, wv, wqkv, NQKV * DM);
  k_swz_w<<<2048, 256, 0, stream>>>(wo, wo, wo, wob, DM * DM);
  k_bias3<<<12, 256, 0, stream>>>(bq, bk, bv, bqkv);

  k_gemm<<<dim3(12, 64), 256, 0, stream>>>(xb, wqkv, bqkv, Qb, Kbf, VbT,
                                           nullptr, 0);
  k_attn<<<dim3(32, 16, 4), 128, 0, stream>>>(Qb, Kbf, VbT, Ab);
  k_gemm<<<dim3(4, 64), 256, 0, stream>>>(Ab, wob, bo, nullptr, nullptr,
                                          nullptr, out, 1);
}
